// Bi_S4A_NS_73418170958501
// MI455X (gfx1250) — matrix operand order repaired and checked by reading, not compiled, not run on hardware
//
#include <hip/hip_runtime.h>
#include <hip/hip_bf16.h>
#include <math.h>
#include <stdint.h>

// ---------------------------------------------------------------- constants
constexpr int Bc  = 2;
constexpr int Lc  = 4096;
constexpr int Dd  = 768;     // model dim
constexpr int BNc = 384;     // bottleneck
constexpr int DIc = 768;     // d_inner
constexpr int NSc = 16;      // d_state
constexpr int DCc = 4;       // d_conv
constexpr int Rc  = 24;      // dt_rank
constexpr int Mr  = Bc * Lc; // 8192 rows
constexpr int XDP = 64;      // x_dbl padded row stride (R + 2*NS = 56 -> 64)
constexpr int TCH = 64;      // scan time-chunk
constexpr int NCH = Lc / TCH;

typedef __attribute__((ext_vector_type(16))) __bf16 v16bf;
typedef __attribute__((ext_vector_type(8)))  __bf16 v8bf;
typedef __attribute__((ext_vector_type(8)))  float  v8f;

__device__ inline float sigmoidf_fast(float x) { return 1.f / (1.f + __expf(-x)); }
__device__ inline float siluf(float x)         { return x * sigmoidf_fast(x); }
__device__ inline float softplusf(float x)     { return x > 20.f ? x : log1pf(__expf(x)); }

// gfx1250 async global->LDS copy (ASYNCcnt-tracked). LDS dest address is the
// low 32 bits of the flat shared-memory pointer (ISA 10.2: LDS aperture maps
// to the low 32 address bits).
__device__ inline void async_copy_b128(void* lds, const float* gaddr) {
  uint32_t l = (uint32_t)(uintptr_t)lds;
  asm volatile("global_load_async_to_lds_b128 %0, %1, off"
               :: "v"(l), "v"(gaddr) : "memory");
}
__device__ inline void wait_async0() {
  asm volatile("s_wait_asynccnt 0x0" ::: "memory");
}

// ---------------------------------------------------------------- WMMA GEMM
// C[M,N] = A[M,K] * B[K,N] (+bias), bf16x3 split precision, double-buffered
// LDS, register prefetch of next K-chunk. Block 256 threads, tile 64x64, BK 32.
// EDGE=false: K%32==0 && N%64==0 -> unconditional loads (no exec masking).
// EDGE=true : ragged K/N handled with clamped addresses + cndmask zeroing.
#define LDSP 40

template <int EPI, bool EDGE> // EPI: 0 none, 1 softplus
__global__ __launch_bounds__(256) void gemm_wmma(
    int M, int N, int K,
    const float* __restrict__ A, int lda,
    const float* __restrict__ Bw, int ldb,
    float* __restrict__ C, int ldc,
    const float* __restrict__ bias)
{
  __shared__ __bf16 Ah[2][64 * LDSP];
  __shared__ __bf16 Al[2][64 * LDSP];
  __shared__ __bf16 Bh[2][64 * LDSP]; // transposed: [n][k]
  __shared__ __bf16 Bl[2][64 * LDSP];

  const int tid  = threadIdx.x;
  const int m0   = blockIdx.y * 64;
  const int n0   = blockIdx.x * 64;
  const int wv   = tid >> 5;
  const int lane = tid & 31;
  const int tr   = wv >> 1;        // 0..3 (16-row tile)
  const int tc0  = (wv & 1) * 2;   // col-tile pair base
  const int m16  = lane & 15;
  const int hi   = lane >> 4;

  // per-thread staging coordinates (fixed across chunks)
  const int aRow[2] = { (tid * 2 + 0) >> 3, (tid * 2 + 1) >> 3 };
  const int aC4[2]  = { ((tid * 2 + 0) & 7) << 2, ((tid * 2 + 1) & 7) << 2 };
  const int bKr[2]  = { (tid * 2 + 0) >> 4, (tid * 2 + 1) >> 4 };
  const int bC4[2]  = { ((tid * 2 + 0) & 15) << 2, ((tid * 2 + 1) & 15) << 2 };

  v8f acc0 = {};
  v8f acc1 = {};
  float4 ra[2], rb[2];

  auto fetch = [&](int k0) {
#pragma unroll
    for (int it = 0; it < 2; ++it) {
      if (EDGE) {
        int kk = k0 + aC4[it];
        bool ok = kk < K;
        int kc = ok ? kk : (K - 4);                 // K%4==0
        float4 v = *(const float4*)(A + (size_t)(m0 + aRow[it]) * lda + kc);
        ra[it] = ok ? v : make_float4(0.f, 0.f, 0.f, 0.f);
        int kr = k0 + bKr[it], nn = n0 + bC4[it];
        bool okb = (kr < K) && (nn < N);
        int krc = kr < K ? kr : (K - 1);
        int nnc = nn < N ? nn : (N - 4);            // N%4==0
        float4 w = *(const float4*)(Bw + (size_t)krc * ldb + nnc);
        rb[it] = okb ? w : make_float4(0.f, 0.f, 0.f, 0.f);
      } else {
        ra[it] = *(const float4*)(A + (size_t)(m0 + aRow[it]) * lda + (k0 + aC4[it]));
        rb[it] = *(const float4*)(Bw + (size_t)(k0 + bKr[it]) * ldb + (n0 + bC4[it]));
      }
    }
  };

  auto stage = [&](int bsel) {
#pragma unroll
    for (int it = 0; it < 2; ++it) {
      float av[4] = { ra[it].x, ra[it].y, ra[it].z, ra[it].w };
      float bv[4] = { rb[it].x, rb[it].y, rb[it].z, rb[it].w };
#pragma unroll
      for (int j = 0; j < 4; ++j) {
        __bf16 hh = (__bf16)av[j];
        Ah[bsel][aRow[it] * LDSP + aC4[it] + j] = hh;
        Al[bsel][aRow[it] * LDSP + aC4[it] + j] = (__bf16)(av[j] - (float)hh);
        __bf16 wh = (__bf16)bv[j];
        Bh[bsel][(bC4[it] + j) * LDSP + bKr[it]] = wh;
        Bl[bsel][(bC4[it] + j) * LDSP + bKr[it]] = (__bf16)(bv[j] - (float)wh);
      }
    }
  };

  const int nch = (K + 31) >> 5;
  fetch(0);

  for (int c = 0; c < nch; ++c) {
    const int bsel = c & 1;
    stage(bsel);
    __syncthreads();
    if (c + 1 < nch) fetch(32 * (c + 1));  // overlap next fetch with WMMAs

    // fragments. A per-lane K: i<8 -> k=i+hi*8 ; i>=8 -> k=i+8+hi*8
    int arow = tr * 16 + m16;
    v8bf ah0 = *(const v8bf*)&Ah[bsel][arow * LDSP + hi * 8];
    v8bf ah1 = *(const v8bf*)&Ah[bsel][arow * LDSP + 16 + hi * 8];
    v8bf al0 = *(const v8bf*)&Al[bsel][arow * LDSP + hi * 8];
    v8bf al1 = *(const v8bf*)&Al[bsel][arow * LDSP + 16 + hi * 8];
    v16bf aH = __builtin_shufflevector(ah0, ah1, 0,1,2,3,4,5,6,7,8,9,10,11,12,13,14,15);
    v16bf aL = __builtin_shufflevector(al0, al1, 0,1,2,3,4,5,6,7,8,9,10,11,12,13,14,15);

#pragma unroll
    for (int ct = 0; ct < 2; ++ct) {
      // B per-lane K: i<8 -> k=i+hi*8 ; i>=8 -> k=i+8+hi*8 (transposed tile)
      int nrow = (tc0 + ct) * 16 + m16;
      v8bf bh0 = *(const v8bf*)&Bh[bsel][nrow * LDSP + hi * 8];
      v8bf bh1 = *(const v8bf*)&Bh[bsel][nrow * LDSP + 16 + hi * 8];
      v8bf bl0 = *(const v8bf*)&Bl[bsel][nrow * LDSP + hi * 8];
      v8bf bl1 = *(const v8bf*)&Bl[bsel][nrow * LDSP + 16 + hi * 8];
      v16bf bH = __builtin_shufflevector(bh0, bh1, 0,1,2,3,4,5,6,7,8,9,10,11,12,13,14,15);
      v16bf bL = __builtin_shufflevector(bl0, bl1, 0,1,2,3,4,5,6,7,8,9,10,11,12,13,14,15);

      v8f& acc = ct ? acc1 : acc0;
      acc = __builtin_amdgcn_wmma_f32_16x16x32_bf16(false, aH, false, bH, (short)0, acc, false, false);
      acc = __builtin_amdgcn_wmma_f32_16x16x32_bf16(false, aH, false, bL, (short)0, acc, false, false);
      acc = __builtin_amdgcn_wmma_f32_16x16x32_bf16(false, aL, false, bH, (short)0, acc, false, false);
    }
    // single barrier per chunk is sufficient with double buffering
  }

  // epilogue: D(m,n): vgpr j -> m = tr*16 + hi*8 + j ; n = tile + (lane&15)
#pragma unroll
  for (int ct = 0; ct < 2; ++ct) {
    v8f acc = ct ? acc1 : acc0;
    int n = n0 + (tc0 + ct) * 16 + m16;
    if (n < N) {
      float bv = bias ? bias[n] : 0.f;
#pragma unroll
      for (int j = 0; j < 8; ++j) {
        int m = m0 + tr * 16 + hi * 8 + j;
        float v = acc[j] + bv;
        if (EPI == 1) v = softplusf(v);
        C[(size_t)m * ldc + n] = v;
      }
    }
  }
}

// ---------------------------------------------------------------- elementwise
__global__ void reverse_rows_kernel(const float* __restrict__ src, float* __restrict__ dst) {
  int idx = blockIdx.x * 256 + threadIdx.x;
  if (idx >= Mr * BNc) return;
  int r = idx / BNc, c = idx - r * BNc;
  int b = r / Lc, t = r - b * Lc;
  dst[idx] = src[(size_t)(b * Lc + (Lc - 1 - t)) * BNc + c];
}

__global__ void conv_silu_kernel(const float* __restrict__ xz, const float* __restrict__ w,
                                 const float* __restrict__ cb, float* __restrict__ u) {
  int idx = blockIdx.x * 256 + threadIdx.x;
  if (idx >= Mr * DIc) return;
  int d = idx % DIc;
  int r = idx / DIc;
  int b = r / Lc, t = r - b * Lc;
  float acc = cb[d];
#pragma unroll
  for (int j = 0; j < DCc; ++j) {
    int tt = t - (DCc - 1) + j;
    if (tt >= 0)
      acc = fmaf(xz[(size_t)(b * Lc + tt) * (2 * DIc) + d], w[d * DCc + j], acc);
  }
  u[idx] = siluf(acc);
}

__global__ void gate_kernel(const float* __restrict__ xz, const float* __restrict__ u,
                            const float* __restrict__ Dp, float* __restrict__ ys) {
  int idx = blockIdx.x * 256 + threadIdx.x;
  if (idx >= Mr * DIc) return;
  int d = idx % DIc;
  int r = idx / DIc;
  float z = xz[(size_t)r * (2 * DIc) + DIc + d];
  ys[idx] = (ys[idx] + u[idx] * Dp[d]) * siluf(z);
}

// ---------------------------------------------------------------- selective scan
// Block = 16 channels x 16 states (one batch b, one direction). Time processed
// in 64-step chunks; dt/u (16 ch) and B/C rows are staged into double-buffered
// LDS by the async global->LDS engine while the previous chunk's recurrence
// runs. Inner loop: 4 LDS broadcast reads + exp/fma + 4 lane shuffles.
__global__ __launch_bounds__(256) void scan_kernel(
    const float* __restrict__ dt0, const float* __restrict__ dt1,
    const float* __restrict__ u0,  const float* __restrict__ u1,
    const float* __restrict__ xd0, const float* __restrict__ xd1,
    const float* __restrict__ Al0, const float* __restrict__ Al1,
    float* __restrict__ ys0, float* __restrict__ ys1)
{
  __shared__ float bcS[2][TCH * 32];  // [t][0..15]=B, [t][16..31]=C
  __shared__ float dtS[2][TCH * 16];
  __shared__ float uS [2][TCH * 16];

  const int dir = blockIdx.y;
  const float* dt = dir ? dt1 : dt0;
  const float* u  = dir ? u1  : u0;
  const float* xd = dir ? xd1 : xd0;
  const float* Al = dir ? Al1 : Al0;
  float*       ys = dir ? ys1 : ys0;

  const int tid = threadIdx.x;
  const int ch0 = blockIdx.x * 16;          // group of 16 channels, one b
  const int dl  = tid >> 4;                 // channel within group
  const int n   = tid & 15;                 // state index
  const int b   = ch0 / DIc;
  const int d   = (ch0 - b * DIc) + dl;

  const float A = -__expf(Al[d * NSc + n]);
  const float* xdp = xd + (size_t)b * Lc * XDP;
  const float* dtp = dt + (size_t)b * Lc * DIc + (ch0 - b * DIc);
  const float* up  = u  + (size_t)b * Lc * DIc + (ch0 - b * DIc);
  float*       ysp = ys + (size_t)b * Lc * DIc + d;

  // async-stage one 64-step chunk: 4 b128 issues per thread
  auto issue = [&](int c, int bsel) {
    int tcb = c * TCH;
#pragma unroll
    for (int i = 0; i < 2; ++i) {
      int slot = tid * 2 + i;               // 512 = 64 rows x 8 xfers
      int tl = slot >> 3, part = slot & 7;
      async_copy_b128(&bcS[bsel][tl * 32 + part * 4],
                      xdp + (size_t)(tcb + tl) * XDP + Rc + part * 4);
    }
    {
      int tl = tid >> 2, part = tid & 3;    // 256 = 64 rows x 4 xfers
      async_copy_b128(&dtS[bsel][tl * 16 + part * 4],
                      dtp + (size_t)(tcb + tl) * DIc + part * 4);
      async_copy_b128(&uS[bsel][tl * 16 + part * 4],
                      up + (size_t)(tcb + tl) * DIc + part * 4);
    }
  };

  float h = 0.f;
  issue(0, 0);
  wait_async0();
  __syncthreads();

  for (int c = 0; c < NCH; ++c) {
    const int bsel = c & 1;
    if (c + 1 < NCH) issue(c + 1, bsel ^ 1);  // overlap transfer with compute

    const float* bcb = bcS[bsel];
    const float* dtb = dtS[bsel];
    const float* ub  = uS[bsel];
    const int tcb = c * TCH;
    for (int tl = 0; tl < TCH; ++tl) {
      float dtv = dtb[tl * 16 + dl];
      float uv  = ub [tl * 16 + dl];
      float Bv  = bcb[tl * 32 + n];
      float Cv  = bcb[tl * 32 + 16 + n];
      float dA = __expf(dtv * A);
      h = fmaf(dA, h, dtv * uv * Bv);
      float p = h * Cv;
      p += __shfl_xor(p, 1);
      p += __shfl_xor(p, 2);
      p += __shfl_xor(p, 4);
      p += __shfl_xor(p, 8);
      if (n == 0) ysp[(size_t)(tcb + tl) * DIc] = p;
    }
    wait_async0();      // own wave's chunk c+1 transfers complete
    __syncthreads();    // everyone's data visible, safe to reuse buffers
  }
}

// ---------------------------------------------------------------- LN + combine
__device__ inline float block_sum_128(float v, float* sm) {
  int t = threadIdx.x;
  sm[t] = v; __syncthreads();
  if (t < 64) sm[t] += sm[t + 64]; __syncthreads();
  if (t < 32) sm[t] += sm[t + 32]; __syncthreads();
  if (t < 16) sm[t] += sm[t + 16]; __syncthreads();
  if (t < 8)  sm[t] += sm[t + 8];  __syncthreads();
  if (t < 4)  sm[t] += sm[t + 4];  __syncthreads();
  if (t < 2)  sm[t] += sm[t + 2];  __syncthreads();
  if (t < 1)  sm[t] += sm[t + 1];  __syncthreads();
  float r = sm[0]; __syncthreads();
  return r;
}

__global__ __launch_bounds__(128) void ln_combine_kernel(
    const float* __restrict__ of, const float* __restrict__ ob,
    const float* __restrict__ gf, const float* __restrict__ bf_,
    const float* __restrict__ gb, const float* __restrict__ bb,
    float* __restrict__ out)
{
  __shared__ float sm[128];
  int row = blockIdx.x;
  int b = row / Lc, t = row - b * Lc;
  int rrev = b * Lc + (Lc - 1 - t);
  const float* pf = of + (size_t)row * BNc;
  const float* pb = ob + (size_t)rrev * BNc;

  float xf[3], xb[3];
  float sf = 0.f, sb = 0.f;
#pragma unroll
  for (int i = 0; i < 3; ++i) {
    int c = threadIdx.x + i * 128;
    xf[i] = pf[c]; xb[i] = pb[c];
    sf += xf[i]; sb += xb[i];
  }
  float mf = block_sum_128(sf, sm) * (1.f / BNc);
  float mb = block_sum_128(sb, sm) * (1.f / BNc);
  float vf = 0.f, vb = 0.f;
#pragma unroll
  for (int i = 0; i < 3; ++i) {
    float d1 = xf[i] - mf; vf += d1 * d1;
    float d2 = xb[i] - mb; vb += d2 * d2;
  }
  vf = block_sum_128(vf, sm) * (1.f / BNc);
  vb = block_sum_128(vb, sm) * (1.f / BNc);
  float rf = rsqrtf(vf + 1e-5f);
  float rb = rsqrtf(vb + 1e-5f);
#pragma unroll
  for (int i = 0; i < 3; ++i) {
    int c = threadIdx.x + i * 128;
    out[(size_t)row * BNc + c] =
        (xf[i] - mf) * rf * gf[c] + bf_[c] + (xb[i] - mb) * rb * gb[c] + bb[c];
  }
}

// ---------------------------------------------------------------- launcher
extern "C" void kernel_launch(void* const* d_in, const int* in_sizes, int n_in,
                              void* d_out, int out_size, void* d_ws, size_t ws_size,
                              hipStream_t stream) {
  (void)in_sizes; (void)n_in; (void)out_size; (void)ws_size;

  const float* x      = (const float*)d_in[0];
  const float* down_W = (const float*)d_in[1];
  const float* down_b = (const float*)d_in[2];
  const float* up_W   = (const float*)d_in[3];
  const float* up_b   = (const float*)d_in[4];

  struct DirP {
    const float *in_W, *conv_w, *conv_b, *xproj_W, *dt_W, *dt_b, *A_log, *Dp, *out_W, *ln_g, *ln_b;
  } P[2];
  for (int p = 0; p < 2; ++p) {
    int o = 5 + p * 11;
    P[p].in_W    = (const float*)d_in[o + 0];
    P[p].conv_w  = (const float*)d_in[o + 1];
    P[p].conv_b  = (const float*)d_in[o + 2];
    P[p].xproj_W = (const float*)d_in[o + 3];
    P[p].dt_W    = (const float*)d_in[o + 4];
    P[p].dt_b    = (const float*)d_in[o + 5];
    P[p].A_log   = (const float*)d_in[o + 6];
    P[p].Dp      = (const float*)d_in[o + 7];
    P[p].out_W   = (const float*)d_in[o + 8];
    P[p].ln_g    = (const float*)d_in[o + 9];
    P[p].ln_b    = (const float*)d_in[o + 10];
  }

  float* ws = (float*)d_ws;
  size_t off = 0;
  auto alloc = [&](size_t n) { float* p = ws + off; off += n; return p; };
  float* h    = alloc((size_t)Mr * BNc);
  float* hrev = alloc((size_t)Mr * BNc);
  float* scmb = alloc((size_t)Mr * BNc);
  float *xzB[2], *uB[2], *xdB[2], *dtB[2], *ysB[2], *oB[2];
  for (int p = 0; p < 2; ++p) {
    xzB[p] = alloc((size_t)Mr * 2 * DIc);
    uB[p]  = alloc((size_t)Mr * DIc);
    xdB[p] = alloc((size_t)Mr * XDP);
    dtB[p] = alloc((size_t)Mr * DIc);
    ysB[p] = alloc((size_t)Mr * DIc);
    oB[p]  = alloc((size_t)Mr * BNc);
  }

  auto gridFor = [](int M, int N) { return dim3((unsigned)((N + 63) / 64), (unsigned)(M / 64)); };

  // 1. h = x @ down_W + down_b
  gemm_wmma<0, false><<<gridFor(Mr, BNc), 256, 0, stream>>>(
      Mr, BNc, Dd, x, Dd, down_W, BNc, h, BNc, down_b);
  // 2. reversed copy for backward direction
  reverse_rows_kernel<<<(Mr * BNc + 255) / 256, 256, 0, stream>>>(h, hrev);

  for (int p = 0; p < 2; ++p) {
    const float* hin = p ? hrev : h;
    // 3. xz = h @ in_W
    gemm_wmma<0, false><<<gridFor(Mr, 2 * DIc), 256, 0, stream>>>(
        Mr, 2 * DIc, BNc, hin, BNc, P[p].in_W, 2 * DIc, xzB[p], 2 * DIc, nullptr);
    // 4. u = silu(conv(xs) + conv_b)
    conv_silu_kernel<<<(Mr * DIc + 255) / 256, 256, 0, stream>>>(xzB[p], P[p].conv_w, P[p].conv_b, uB[p]);
    // 5. x_dbl = u @ xproj_W   (ragged N=56 into padded stride 64)
    gemm_wmma<0, true><<<gridFor(Mr, Rc + 2 * NSc), 256, 0, stream>>>(
        Mr, Rc + 2 * NSc, DIc, uB[p], DIc, P[p].xproj_W, Rc + 2 * NSc, xdB[p], XDP, nullptr);
    // 6. dt = softplus(x_dbl[:, :R] @ dt_W + dt_b)   (ragged K=24, lda=64)
    gemm_wmma<1, true><<<gridFor(Mr, DIc), 256, 0, stream>>>(
        Mr, DIc, Rc, xdB[p], XDP, P[p].dt_W, DIc, dtB[p], DIc, P[p].dt_b);
  }

  // 7. fused bidirectional selective scan (async-LDS staged)
  scan_kernel<<<dim3((unsigned)(Bc * DIc / 16), 2), 256, 0, stream>>>(
      dtB[0], dtB[1], uB[0], uB[1], xdB[0], xdB[1], P[0].A_log, P[1].A_log, ysB[0], ysB[1]);

  for (int p = 0; p < 2; ++p) {
    // 8. ys = (ys + u*D) * silu(z)
    gate_kernel<<<(Mr * DIc + 255) / 256, 256, 0, stream>>>(xzB[p], uB[p], P[p].Dp, ysB[p]);
    // 9. o = ys @ out_W
    gemm_wmma<0, false><<<gridFor(Mr, BNc), 256, 0, stream>>>(
        Mr, BNc, DIc, ysB[p], DIc, P[p].out_W, BNc, oB[p], BNc, nullptr);
  }

  // 10. s = LN(o_f) + reverse(LN(o_b))
  ln_combine_kernel<<<Mr, 128, 0, stream>>>(
      oB[0], oB[1], P[0].ln_g, P[0].ln_b, P[1].ln_g, P[1].ln_b, scmb);

  // 11. out = s @ up_W + up_b
  gemm_wmma<0, false><<<gridFor(Mr, Dd), 256, 0, stream>>>(
      Mr, Dd, BNc, scmb, BNc, up_W, Dd, (float*)d_out, Dd, up_b);
}
